// MultiHeadAttention_24300924961445
// MI455X (gfx1250) — compile-verified, matrix operand order checked by reading, not run on hardware
//
#include <hip/hip_runtime.h>
#include <hip/hip_bf16.h>

typedef __attribute__((ext_vector_type(16))) _Float16 v16h;
typedef __attribute__((ext_vector_type(8)))  _Float16 v8h;
typedef __attribute__((ext_vector_type(4)))  _Float16 v4h;
typedef __attribute__((ext_vector_type(8)))  float    v8f;
typedef __attribute__((ext_vector_type(4)))  float    v4f;

#define D_MODEL 1024
#define SEQ     4096
#define NHEAD   16
#define DK      64

// ---------------------------------------------------------------------------
// helpers
// ---------------------------------------------------------------------------
static __device__ __forceinline__ v8f zero8() {
  v8f z;
#pragma unroll
  for (int i = 0; i < 8; ++i) z[i] = 0.0f;
  return z;
}

// Pure-VALU lane permute within DPP rows (no LDS round trip).
template <int CTRL>
static __device__ __forceinline__ float dpp_f(float x) {
  int i = __float_as_int(x);
  i = __builtin_amdgcn_mov_dpp(i, CTRL, 0xf, 0xf, true);
  return __int_as_float(i);
}

// Reduce across each 16-lane group (a DPP "row") with a VALU butterfly:
// xor1, xor2 via quad_perm; after quads are uniform, row_half_mirror == xor4
// and row_mirror == xor8.
static __device__ __forceinline__ float red16_max(float v) {
  v = fmaxf(v, dpp_f<0xB1>(v));   // quad_perm [1,0,3,2]  (xor 1)
  v = fmaxf(v, dpp_f<0x4E>(v));   // quad_perm [2,3,0,1]  (xor 2)
  v = fmaxf(v, dpp_f<0x141>(v));  // row_half_mirror      (== xor 4)
  v = fmaxf(v, dpp_f<0x140>(v));  // row_mirror           (== xor 8)
  return v;
}
static __device__ __forceinline__ float red16_sum(float v) {
  v += dpp_f<0xB1>(v);
  v += dpp_f<0x4E>(v);
  v += dpp_f<0x141>(v);
  v += dpp_f<0x140>(v);
  return v;
}

// Load one 16-lane-striped f16 WMMA fragment (A or B) from a row whose
// reduction dim is contiguous.  p = &row[0] within a 32-wide K window,
// khi = (lane>>4)*8.  Per the CDNA5 16-bit layout, the lane's 16 halves are
// K = {khi..khi+7} and {16+khi..16+khi+7}: two aligned 16-byte loads.
static __device__ __forceinline__ v16h ld_frag(const _Float16* p, int khi) {
  v8h lo = *(const v8h*)(p + khi);
  v8h hi = *(const v8h*)(p + 16 + khi);
  return __builtin_shufflevector(lo, hi, 0, 1, 2, 3, 4, 5, 6, 7,
                                         8, 9, 10, 11, 12, 13, 14, 15);
}

// ---------------------------------------------------------------------------
// GEMM:  Out[M x N] = (A[M x K] @ W[N x K]^T + bias) * oscale   (torch Linear)
// Block tile 128x128, K-tile 32, 256 threads = 8 waves, wave tile 32x64.
// A_HALF: A is f16 (else f32).  OUT_HALF: store f16 (else f32).
// ---------------------------------------------------------------------------
template <bool A_HALF, bool OUT_HALF>
__global__ __launch_bounds__(256) void gemm_bias_kernel(
    const void* __restrict__ Ain, const float* __restrict__ W,
    const float* __restrict__ bias, void* __restrict__ Out,
    int K, int N, float oscale) {
  constexpr int LDT = 40;  // LDS row stride in halves (80B, 16B aligned, padded)
  __shared__ _Float16 As[128 * LDT];
  __shared__ _Float16 Ws[128 * LDT];

  const int tid   = threadIdx.x;
  const int lane  = tid & 31;
  const int wave  = tid >> 5;
  const int mBase = blockIdx.y * 128;
  const int nBase = blockIdx.x * 128;
  const int waveM = (wave >> 1) * 32;  // 4 waves down M
  const int waveN = (wave & 1) * 64;   // 2 waves across N
  const int khi   = (lane >> 4) * 8;
  const int lr    = lane & 15;

  v8f acc[2][4];
#pragma unroll
  for (int i = 0; i < 2; ++i)
#pragma unroll
    for (int j = 0; j < 4; ++j) acc[i][j] = zero8();

  const int row = tid >> 1;         // 0..127
  const int seg = (tid & 1) * 16;   // 0 or 16 (halves / floats)

  for (int k0 = 0; k0 < K; k0 += 32) {
    __syncthreads();
    // ---- stage A tile (convert f32 -> f16 if needed) ----
    if (A_HALF) {
      const _Float16* gA =
          (const _Float16*)Ain + (size_t)(mBase + row) * K + k0 + seg;
      *(v8h*)&As[row * LDT + seg]     = *(const v8h*)gA;
      *(v8h*)&As[row * LDT + seg + 8] = *(const v8h*)(gA + 8);
      if (k0 + 32 < K) __builtin_prefetch(gA + 32, 0, 3);
    } else {
      const float* gA = (const float*)Ain + (size_t)(mBase + row) * K + k0 + seg;
      v4f f0 = *(const v4f*)gA;
      v4f f1 = *(const v4f*)(gA + 4);
      v4f f2 = *(const v4f*)(gA + 8);
      v4f f3 = *(const v4f*)(gA + 12);
      v8h h0, h1;
#pragma unroll
      for (int i = 0; i < 4; ++i) {
        h0[i]     = (_Float16)f0[i];
        h0[i + 4] = (_Float16)f1[i];
        h1[i]     = (_Float16)f2[i];
        h1[i + 4] = (_Float16)f3[i];
      }
      *(v8h*)&As[row * LDT + seg]     = h0;
      *(v8h*)&As[row * LDT + seg + 8] = h1;
      if (k0 + 32 < K) __builtin_prefetch(gA + 32, 0, 3);
    }
    // ---- stage W tile: Ws[n][k] = W[nBase+n][k0+k]  (f32 -> f16) ----
    {
      const float* gW = W + (size_t)(nBase + row) * K + k0 + seg;
      v4f f0 = *(const v4f*)gW;
      v4f f1 = *(const v4f*)(gW + 4);
      v4f f2 = *(const v4f*)(gW + 8);
      v4f f3 = *(const v4f*)(gW + 12);
      v8h h0, h1;
#pragma unroll
      for (int i = 0; i < 4; ++i) {
        h0[i]     = (_Float16)f0[i];
        h0[i + 4] = (_Float16)f1[i];
        h1[i]     = (_Float16)f2[i];
        h1[i + 4] = (_Float16)f3[i];
      }
      *(v8h*)&Ws[row * LDT + seg]     = h0;
      *(v8h*)&Ws[row * LDT + seg + 8] = h1;
      if (k0 + 32 < K) __builtin_prefetch(gW + 32, 0, 3);
    }
    __syncthreads();

    // ---- one 32-deep WMMA K-step over the 32x64 wave tile ----
    v16h a[2], b[4];
#pragma unroll
    for (int i = 0; i < 2; ++i)
      a[i] = ld_frag(&As[(waveM + i * 16 + lr) * LDT], khi);
#pragma unroll
    for (int j = 0; j < 4; ++j)
      b[j] = ld_frag(&Ws[(waveN + j * 16 + lr) * LDT], khi);
#pragma unroll
    for (int i = 0; i < 2; ++i)
#pragma unroll
      for (int j = 0; j < 4; ++j)
        acc[i][j] = __builtin_amdgcn_wmma_f32_16x16x32_f16(
            false, a[i], false, b[j], (short)0, acc[i][j], false, false);
  }

  // ---- epilogue: (+ bias) * oscale, store ----
#pragma unroll
  for (int j = 0; j < 4; ++j) {
    const int n    = nBase + waveN + j * 16 + lr;
    const float bj = bias[n];
#pragma unroll
    for (int i = 0; i < 2; ++i) {
#pragma unroll
      for (int r = 0; r < 8; ++r) {
        const int m = mBase + waveM + i * 16 + r + ((lane >> 4) * 8);
        const float v = (acc[i][j][r] + bj) * oscale;
        if (OUT_HALF)
          ((_Float16*)Out)[(size_t)m * N + n] = (_Float16)v;
        else
          ((float*)Out)[(size_t)m * N + n] = v;
      }
    }
  }
}

// ---------------------------------------------------------------------------
// Flash attention, one block per (64 q-rows, head).  128 threads = 4 waves,
// each wave owns 16 q-rows.  Iterates over all 4096 keys in 64-key tiles.
// Q arrives pre-scaled by 1/sqrt(dk).
// ---------------------------------------------------------------------------
__global__ __launch_bounds__(128) void flash_attn_kernel(
    const _Float16* __restrict__ Q, const _Float16* __restrict__ Kh,
    const _Float16* __restrict__ Vh, _Float16* __restrict__ O) {
  constexpr int LDK = 72;  // 144B rows: 16B aligned, 8-bank padded
  __shared__ _Float16 Ks[64 * LDK];      // [key][d]      (row major)
  __shared__ _Float16 Vt[64 * LDK];      // [d][key]      (transposed)
  __shared__ _Float16 Ps[4 * 16 * LDK];  // per-wave P scratch [m][key]

  const int tid   = threadIdx.x;
  const int lane  = tid & 31;
  const int wave  = tid >> 5;
  const int head  = blockIdx.y;
  const int qBase = blockIdx.x * 64 + wave * 16;
  const int hcol  = head * DK;
  const int khi   = (lane >> 4) * 8;
  const int lr    = lane & 15;

  // Q fragments for this wave's 16 rows: two 32-deep K-steps over d=64
  v16h aq[2];
#pragma unroll
  for (int ks = 0; ks < 2; ++ks)
    aq[ks] = ld_frag(Q + (size_t)(qBase + lr) * D_MODEL + hcol + ks * 32, khi);

  v8f o[4];
#pragma unroll
  for (int t = 0; t < 4; ++t) o[t] = zero8();
  float mrow[8], lrow[8];
#pragma unroll
  for (int r = 0; r < 8; ++r) { mrow[r] = -1e30f; lrow[r] = 0.0f; }

  // K staging: thread owns 1 key row x 32 halves
  const int krow = tid >> 1;        // 0..63
  const int kseg = (tid & 1) * 32;  // 0 or 32 halves along d
  // V staging (transpose): thread owns 4 key rows x 8 halves -> packed b64
  const int vrow = (tid >> 3) * 4;  // 0,4,..,60
  const int vseg = (tid & 7) * 8;   // 0..56 halves along d

  for (int j = 0; j < SEQ; j += 64) {
    __syncthreads();  // previous iteration's tile reads complete
    // ---- stage K tile (row major) ----
    {
      const _Float16* gk = Kh + (size_t)(j + krow) * D_MODEL + hcol + kseg;
#pragma unroll
      for (int c = 0; c < 4; ++c)
        *(v8h*)&Ks[krow * LDK + kseg + c * 8] = *(const v8h*)(gk + c * 8);
      if (j + 64 < SEQ) __builtin_prefetch(gk + (size_t)64 * D_MODEL, 0, 3);
    }
    // ---- stage V tile transposed: Vt[d][key], packed 4-key b64 stores ----
    {
      v8h vr[4];
#pragma unroll
      for (int rr = 0; rr < 4; ++rr)
        vr[rr] =
            *(const v8h*)(Vh + (size_t)(j + vrow + rr) * D_MODEL + hcol + vseg);
#pragma unroll
      for (int d = 0; d < 8; ++d) {
        v4h pk;
#pragma unroll
        for (int rr = 0; rr < 4; ++rr) pk[rr] = vr[rr][d];
        *(v4h*)&Vt[(vseg + d) * LDK + vrow] = pk;
      }
      if (j + 64 < SEQ)
        __builtin_prefetch(
            Vh + (size_t)(j + 64 + vrow) * D_MODEL + hcol + vseg, 0, 3);
    }
    __syncthreads();

    // ---- S = Q K^T (Q pre-scaled): 16x64 per wave, 8 WMMA ----
    v8f s[4];
#pragma unroll
    for (int t = 0; t < 4; ++t) {
      v8f c = zero8();
#pragma unroll
      for (int ks = 0; ks < 2; ++ks) {
        v16h b = ld_frag(&Ks[(t * 16 + lr) * LDK + ks * 32], khi);
        c = __builtin_amdgcn_wmma_f32_16x16x32_f16(false, aq[ks], false, b,
                                                   (short)0, c, false, false);
      }
      s[t] = c;
    }

    // ---- online softmax (rows live across 16-lane groups; VALU DPP) ----
    float mnew[8], alpha[8];
#pragma unroll
    for (int r = 0; r < 8; ++r) {
      float mr = fmaxf(fmaxf(s[0][r], s[1][r]), fmaxf(s[2][r], s[3][r]));
      mr = red16_max(mr);
      const float mn = fmaxf(mrow[r], mr);
      alpha[r] = __expf(mrow[r] - mn);
      mrow[r]  = mn;
      mnew[r]  = mn;
    }
    float rs[8];
#pragma unroll
    for (int r = 0; r < 8; ++r) rs[r] = 0.0f;
#pragma unroll
    for (int t = 0; t < 4; ++t)
#pragma unroll
      for (int r = 0; r < 8; ++r) {
        const float p = __expf(s[t][r] - mnew[r]);
        s[t][r] = p;
        rs[r] += p;
      }
#pragma unroll
    for (int r = 0; r < 8; ++r) {
      lrow[r] = lrow[r] * alpha[r] + red16_sum(rs[r]);
    }
#pragma unroll
    for (int t = 0; t < 4; ++t)
#pragma unroll
      for (int r = 0; r < 8; ++r) o[t][r] *= alpha[r];

    // ---- C-layout -> A-layout via per-wave LDS slab (in-order per wave) ----
    _Float16* Pw = &Ps[wave * 16 * LDK];
#pragma unroll
    for (int t = 0; t < 4; ++t)
#pragma unroll
      for (int r = 0; r < 8; ++r)
        Pw[(r + ((lane >> 4) * 8)) * LDK + t * 16 + lr] = (_Float16)s[t][r];

    v16h ap[2];
#pragma unroll
    for (int ks = 0; ks < 2; ++ks)
      ap[ks] = ld_frag(&Pw[lr * LDK + ks * 32], khi);

    // ---- O += P @ V: 8 WMMA (B frags read contiguous rows of V^T) ----
#pragma unroll
    for (int t = 0; t < 4; ++t) {
#pragma unroll
      for (int ks = 0; ks < 2; ++ks) {
        v16h b = ld_frag(&Vt[(t * 16 + lr) * LDK + ks * 32], khi);
        o[t] = __builtin_amdgcn_wmma_f32_16x16x32_f16(false, ap[ks], false, b,
                                                      (short)0, o[t], false,
                                                      false);
      }
    }
  }

  // ---- normalize and store (f16, merged-head layout [seq][d_model]) ----
#pragma unroll
  for (int r = 0; r < 8; ++r) lrow[r] = 1.0f / lrow[r];
#pragma unroll
  for (int t = 0; t < 4; ++t)
#pragma unroll
    for (int r = 0; r < 8; ++r) {
      const int m = qBase + r + ((lane >> 4) * 8);
      O[(size_t)m * D_MODEL + hcol + t * 16 + lr] =
          (_Float16)(o[t][r] * lrow[r]);
    }
}

// ---------------------------------------------------------------------------
// launch
// ---------------------------------------------------------------------------
extern "C" void kernel_launch(void* const* d_in, const int* in_sizes, int n_in,
                              void* d_out, int out_size, void* d_ws,
                              size_t ws_size, hipStream_t stream) {
  const float* query = (const float*)d_in[0];
  const float* key   = (const float*)d_in[1];
  const float* value = (const float*)d_in[2];
  const float* Wq = (const float*)d_in[3];
  const float* bq = (const float*)d_in[4];
  const float* Wk = (const float*)d_in[5];
  const float* bk = (const float*)d_in[6];
  const float* Wv = (const float*)d_in[7];
  const float* bv = (const float*)d_in[8];
  const float* Wo = (const float*)d_in[9];
  const float* bo = (const float*)d_in[10];
  float* out = (float*)d_out;

  const size_t mat = (size_t)SEQ * D_MODEL;
  _Float16* Qh = (_Float16*)d_ws;
  _Float16* Kh = Qh + mat;
  _Float16* Vh = Kh + mat;
  _Float16* Ah = Vh + mat;  // total 32 MB of f16 scratch

  dim3 gGemm(D_MODEL / 128, SEQ / 128);
  // Q projection folds in the 1/sqrt(dk) softmax scale.
  gemm_bias_kernel<false, true><<<gGemm, 256, 0, stream>>>(
      query, Wq, bq, Qh, D_MODEL, D_MODEL, 0.125f);
  gemm_bias_kernel<false, true><<<gGemm, 256, 0, stream>>>(
      key, Wk, bk, Kh, D_MODEL, D_MODEL, 1.0f);
  gemm_bias_kernel<false, true><<<gGemm, 256, 0, stream>>>(
      value, Wv, bv, Vh, D_MODEL, D_MODEL, 1.0f);
  flash_attn_kernel<<<dim3(SEQ / 64, NHEAD), 128, 0, stream>>>(Qh, Kh, Vh, Ah);
  gemm_bias_kernel<true, false><<<gGemm, 256, 0, stream>>>(
      Ah, Wo, bo, out, D_MODEL, D_MODEL, 1.0f);
}
